// CrossAttention_73890617360614
// MI455X (gfx1250) — matrix operand order repaired and checked by reading, not compiled, not run on hardware
//
#include <hip/hip_runtime.h>
#include <hip/hip_bf16.h>

// ---------------------------------------------------------------------------
// CDNA5 (gfx1250) wave32 WMMA types
// ---------------------------------------------------------------------------
typedef __bf16 bf16;
typedef __attribute__((ext_vector_type(16))) __bf16 v16bf;
typedef __attribute__((ext_vector_type(8)))  float  v8f;

__device__ __forceinline__ v8f wmma_bf16(v16bf a, v16bf b, v8f c) {
  // D = A(16x32 bf16) * B(32x16 bf16) + C(16x16 f32)
  return __builtin_amdgcn_wmma_f32_16x16x32_bf16(false, a, false, b,
                                                 (short)0, c, false, false);
}

// monotonic float<->uint mapping for order-preserving integer compares
__device__ __forceinline__ unsigned mono(float f) {
  unsigned u = __float_as_uint(f);
  return (u & 0x80000000u) ? ~u : (u | 0x80000000u);
}
__device__ __forceinline__ float imono(unsigned u) {
  return (u & 0x80000000u) ? __uint_as_float(u & 0x7FFFFFFFu)
                           : __uint_as_float(~u);
}

#define NEGF (-3.402823466e38f)   // matches jnp.finfo(f32).min

// ---------------------------------------------------------------------------
// Generic 64x64-tile GEMM:  Out = A(MxK) @ B(KxN) + bias
// MODE 0: f32 row-major output (final projection)
// MODE 1: bf16 head-split output  (B,H,T,64)   (Q/K/V projections)
// ABF16: A operand stored as bf16 (true) or f32 (false)
// 256 threads = 8 waves; each wave owns two 16x16 WMMA sub-tiles.
// LDS layouts chosen so every fragment load is a contiguous, 16B-aligned
// run -> ds_load_b128 (A: 2x16B runs/lane, B^T: 2x16B runs/lane).
// ---------------------------------------------------------------------------
template <int MODE, bool ABF16>
__global__ __launch_bounds__(256) void gemm_kernel(
    const void* __restrict__ Aptr, const float* __restrict__ Bptr,
    const float* __restrict__ bias, void* __restrict__ Out,
    int M, int N, int K, int T) {
  __shared__ alignas(16) __bf16 As[64][40];   // 80B row stride: 16B aligned
  __shared__ alignas(16) __bf16 BsT[64][40];  // B stored transposed [col][k]

  const int tid  = threadIdx.x;
  const int lane = tid & 31;
  const int wave = tid >> 5;
  const int hi   = lane >> 4;      // 0 for lanes 0-15, 1 for lanes 16-31
  const int lo   = lane & 15;

  const int r0 = blockIdx.x * 64;
  const int c0 = blockIdx.y * 64;
  const int si  = wave & 3;        // sub-tile row   (0..3)
  const int sj0 = (wave >> 2) * 2; // sub-tile col base (0 or 2)

  const float* Af = (const float*)Aptr;
  const __bf16* Ab = (const __bf16*)Aptr;

  v8f acc0 = {}, acc1 = {};

  for (int kk = 0; kk < K; kk += 32) {
    // ---- stage A 64x32 (coalesced global reads) ----
#pragma unroll
    for (int e = 0; e < 8; ++e) {
      int idx = e * 256 + tid;
      int r = idx >> 5, c = idx & 31;
      float v = ABF16 ? (float)Ab[(size_t)(r0 + r) * K + kk + c]
                      : Af[(size_t)(r0 + r) * K + kk + c];
      As[r][c] = (__bf16)v;
    }
    // ---- stage B 32x64 transposed (coalesced global reads) ----
#pragma unroll
    for (int e = 0; e < 8; ++e) {
      int idx = e * 256 + tid;
      int r = idx >> 6, c = idx & 63;
      BsT[c][r] = (__bf16)Bptr[(size_t)(kk + r) * N + c0 + c];
    }
    // ---- prefetch next k-tile into cache (global_prefetch_b8) ----
    if (kk + 32 < K) {
      int ra = tid >> 5, ca = tid & 31;
      if (ABF16)
        __builtin_prefetch(&Ab[(size_t)(r0 + ra) * K + kk + 32 + ca], 0, 1);
      else
        __builtin_prefetch(&Af[(size_t)(r0 + ra) * K + kk + 32 + ca], 0, 1);
      int rb = tid >> 6, cb = tid & 63;
      __builtin_prefetch(&Bptr[(size_t)(kk + 32 + rb) * N + c0 + cb], 0, 1);
    }
    __syncthreads();

    // ---- build fragments per documented gfx1250 WMMA layouts ----
    v16bf afrag, bfrag0, bfrag1;
#pragma unroll
    for (int t = 0; t < 16; ++t) {
      int k = (t & 7) + (hi << 3) + ((t >> 3) << 4);  // A 16x32 bf16 layout
      afrag[t] = As[si * 16 + lo][k];
    }
#pragma unroll
    for (int t = 0; t < 16; ++t) {
      int k = (t & 7) + (hi << 3) + ((t >> 3) << 4);  // B 32x16 bf16 layout
      bfrag0[t] = BsT[sj0 * 16 + lo][k];
      bfrag1[t] = BsT[(sj0 + 1) * 16 + lo][k];
    }
    acc0 = wmma_bf16(afrag, bfrag0, acc0);
    acc1 = wmma_bf16(afrag, bfrag1, acc1);
    __syncthreads();
  }

  // ---- epilogue: C layout = VGPR v -> M = v + 8*hi, N = lane&15 ----
#pragma unroll
  for (int v = 0; v < 8; ++v) {
    int m = v + (hi << 3);
    int r = r0 + si * 16 + m;
#pragma unroll
    for (int jj = 0; jj < 2; ++jj) {
      int c = c0 + (sj0 + jj) * 16 + lo;
      float val = (jj ? acc1[v] : acc0[v]) + bias[c];
      if (MODE == 0) {
        ((float*)Out)[(size_t)r * N + c] = val;
      } else {
        int bb = r / T, t = r % T;
        int hh = c >> 6, d = c & 63;
        ((__bf16*)Out)[((((size_t)bb * 16 + hh) * T + t) << 6) + d] =
            (__bf16)val;
      }
    }
  }
}

// ---------------------------------------------------------------------------
// Fused attention: one block per (b, h, 16-query tile). 8 waves.
//  phase 1: logits(16x2048) = Q*K^T * scale (+mask)  -> LDS (never HBM)
//  phase 2: exact 64th-largest per row (radix binary search on mono bits),
//           softmax weights written back to LDS
//  phase 3: O(16x64) = P*V via WMMA, key-range split across waves,
//           ds_add_f32 reduction, bf16 output in (B*TQ, H*64) layout
// ---------------------------------------------------------------------------
#define LSTRIDE 2052   // logits row stride (words), breaks bank aliasing

__global__ __launch_bounds__(256) void attn_kernel(
    const __bf16* __restrict__ Q,   // (B,H,1024,64)
    const __bf16* __restrict__ Kh_, // (B,H,2048,64)
    const __bf16* __restrict__ Vh_, // (B,H,2048,64)
    const unsigned char* __restrict__ kmask,  // (B,2048) nonzero = masked
    __bf16* __restrict__ O) {       // (B*1024, 1024)
  extern __shared__ char smem[];
  float* logitsS = (float*)smem;                 // 16 * LSTRIDE
  float* otile   = logitsS + 16 * LSTRIDE;       // 16 * 64
  float* invsum  = otile + 16 * 64;              // 16

  const int tid  = threadIdx.x;
  const int lane = tid & 31;
  const int wave = tid >> 5;
  const int hi   = lane >> 4;
  const int lo   = lane & 15;

  const int qt = blockIdx.x, h = blockIdx.y, b = blockIdx.z;
  const int q0 = qt * 16;
  const size_t bh = (size_t)b * 16 + h;
  const __bf16* Qh = Q   + bh * 1024 * 64;
  const __bf16* Kp = Kh_ + bh * 2048 * 64;
  const __bf16* Vp = Vh_ + bh * 2048 * 64;
  const unsigned char* mrow = kmask + (size_t)b * 2048;

  for (int e = tid; e < 1024; e += 256) otile[e] = 0.0f;

  // preload Q fragments for the two K=32 steps covering DH=64
  v16bf aq0, aq1;
#pragma unroll
  for (int t = 0; t < 16; ++t) {
    int k = (t & 7) + (hi << 3) + ((t >> 3) << 4);
    aq0[t] = Qh[(size_t)(q0 + lo) * 64 + k];
    aq1[t] = Qh[(size_t)(q0 + lo) * 64 + 32 + k];
  }

  // ---- phase 1: logits ----
  for (int kt = wave; kt < 128; kt += 8) {
    v16bf bk0, bk1;
#pragma unroll
    for (int t = 0; t < 16; ++t) {
      int k = (t & 7) + (hi << 3) + ((t >> 3) << 4);
      bk0[t] = Kp[(size_t)(kt * 16 + lo) * 64 + k];
      bk1[t] = Kp[(size_t)(kt * 16 + lo) * 64 + 32 + k];
    }
    // prefetch next key tile for this wave
    if (kt + 8 < 128)
      __builtin_prefetch(&Kp[(size_t)((kt + 8) * 16 + lo) * 64 + (hi << 4)],
                         0, 1);
    v8f acc = {};
    acc = wmma_bf16(aq0, bk0, acc);
    acc = wmma_bf16(aq1, bk1, acc);

    int key = kt * 16 + lo;
    bool masked = mrow[key] != 0;
#pragma unroll
    for (int v = 0; v < 8; ++v) {
      float l = acc[v] * 0.125f;          // 1/sqrt(DH=64)
      if (masked) l = NEGF;
      logitsS[(v + (hi << 3)) * LSTRIDE + key] = l;
    }
  }
  __syncthreads();

  // ---- phase 2: top-64 threshold + softmax weights (2 rows per wave) ----
  for (int rr = 0; rr < 2; ++rr) {
    int row = wave * 2 + rr;
    unsigned u[64];
#pragma unroll
    for (int i = 0; i < 64; ++i)
      u[i] = mono(logitsS[row * LSTRIDE + i * 32 + lane]);

    // radix binary search: largest t with count(x >= t) >= 64  == 64th largest
    unsigned cand = 0;
    for (int bit = 31; bit >= 0; --bit) {
      unsigned trial = cand | (1u << bit);
      int cnt = 0;
#pragma unroll
      for (int i = 0; i < 64; ++i) cnt += (u[i] >= trial) ? 1 : 0;
      for (int off = 16; off; off >>= 1) cnt += __shfl_xor(cnt, off);
      if (cnt >= 64) cand = trial;
    }

    unsigned um = 0;
#pragma unroll
    for (int i = 0; i < 64; ++i) um = (u[i] > um) ? u[i] : um;
    for (int off = 16; off; off >>= 1) {
      unsigned o = __shfl_xor(um, off);
      um = (o > um) ? o : um;
    }
    float mx = imono(um);

    float s = 0.0f;
#pragma unroll
    for (int i = 0; i < 64; ++i) {
      float w = 0.0f;
      if (u[i] >= cand) w = expf(imono(u[i]) - mx);
      s += w;
      logitsS[row * LSTRIDE + i * 32 + lane] = w;
    }
    for (int off = 16; off; off >>= 1) s += __shfl_xor(s, off);
    if (lane == 0) invsum[row] = 1.0f / s;
  }
  __syncthreads();

  // ---- phase 3: O = P * V, keys split across waves ----
  v8f oacc[4] = {{}, {}, {}, {}};
  const int ks0 = wave * 256;
  for (int ks = ks0; ks < ks0 + 256; ks += 32) {
    v16bf aw;
#pragma unroll
    for (int t = 0; t < 16; ++t) {
      int k = (t & 7) + (hi << 3) + ((t >> 3) << 4);
      aw[t] = (__bf16)logitsS[lo * LSTRIDE + ks + k];
    }
#pragma unroll
    for (int d = 0; d < 4; ++d) {
      v16bf bv;
#pragma unroll
      for (int t = 0; t < 16; ++t) {
        int k = (t & 7) + (hi << 3) + ((t >> 3) << 4);
        bv[t] = Vp[(size_t)(ks + k) * 64 + d * 16 + lo];
      }
      oacc[d] = wmma_bf16(aw, bv, oacc[d]);
    }
  }
#pragma unroll
  for (int d = 0; d < 4; ++d)
#pragma unroll
    for (int v = 0; v < 8; ++v) {
      int m = v + (hi << 3);
      atomicAdd(&otile[m * 64 + d * 16 + lo], oacc[d][v]);
    }
  __syncthreads();

  for (int e = tid; e < 1024; e += 256) {
    int m = e >> 6, d = e & 63;
    float val = otile[e] * invsum[m];
    O[(size_t)(b * 1024 + q0 + m) * 1024 + h * 64 + d] = (__bf16)val;
  }
}

// ---------------------------------------------------------------------------
// Host launcher
// ---------------------------------------------------------------------------
extern "C" void kernel_launch(void* const* d_in, const int* in_sizes, int n_in,
                              void* d_out, int out_size, void* d_ws,
                              size_t ws_size, hipStream_t stream) {
  (void)in_sizes; (void)n_in; (void)out_size; (void)ws_size;
  const float* x    = (const float*)d_in[0];   // (2,1024,1024)
  const float* ctx  = (const float*)d_in[1];   // (2,2048,768)
  const unsigned char* mask = (const unsigned char*)d_in[2]; // (2,2048) bool
  const float* Wq = (const float*)d_in[3];
  const float* bq = (const float*)d_in[4];
  const float* Wk = (const float*)d_in[5];
  const float* bk = (const float*)d_in[6];
  const float* Wv = (const float*)d_in[7];
  const float* bv = (const float*)d_in[8];
  const float* Wo = (const float*)d_in[9];
  const float* bo = (const float*)d_in[10];

  char* ws = (char*)d_ws;
  __bf16* Qws = (__bf16*)(ws);                        //  4 MB (2,16,1024,64)
  __bf16* Kws = (__bf16*)(ws + (size_t)4  * 1048576); //  8 MB (2,16,2048,64)
  __bf16* Vws = (__bf16*)(ws + (size_t)12 * 1048576); //  8 MB
  __bf16* Ows = (__bf16*)(ws + (size_t)20 * 1048576); //  4 MB (2048,1024)

  dim3 blk(256);
  // Q = x @ Wq + bq  -> head-split bf16
  gemm_kernel<1, false><<<dim3(2048 / 64, 1024 / 64), blk, 0, stream>>>(
      x, Wq, bq, Qws, 2048, 1024, 1024, 1024);
  // K = ctx @ Wk + bk
  gemm_kernel<1, false><<<dim3(4096 / 64, 1024 / 64), blk, 0, stream>>>(
      ctx, Wk, bk, Kws, 4096, 1024, 768, 2048);
  // V = ctx @ Wv + bv
  gemm_kernel<1, false><<<dim3(4096 / 64, 1024 / 64), blk, 0, stream>>>(
      ctx, Wv, bv, Vws, 4096, 1024, 768, 2048);

  // fused top-k attention: grid (q-tiles, heads, batch)
  size_t smem = (size_t)(16 * LSTRIDE + 16 * 64 + 16) * sizeof(float);
  attn_kernel<<<dim3(64, 16, 2), blk, smem, stream>>>(Qws, Kws, Vws, mask,
                                                      Ows);

  // out = O @ Wo + bo   (f32 output)
  gemm_kernel<0, true><<<dim3(2048 / 64, 1024 / 64), blk, 0, stream>>>(
      Ows, Wo, bo, (float*)d_out, 2048, 1024, 1024, 1);
}
